// Flux2Attention_75041668596519
// MI455X (gfx1250) — matrix operand order repaired and checked by reading, not compiled, not run on hardware
//
#include <hip/hip_runtime.h>
#include <hip/hip_bf16.h>
#include <stdint.h>

// ---------------------------------------------------------------------------
// Flux2 joint attention for gfx1250 (MI455X), bf16 WMMA everywhere.
// Shapes: B=1, S_IMG=2048, S_TXT=512, S=2560, D=3072, H=24, DH=128.
// ---------------------------------------------------------------------------

#define S_IMG_N 2048
#define S_TXT_N 512
#define S_TOT_N 2560
#define D_N     3072
#define H_N     24
#define DH_N    128

typedef __attribute__((ext_vector_type(16))) __bf16 bf16x16;
typedef __attribute__((ext_vector_type(8)))  float  f32x8;

__device__ __forceinline__ unsigned short f2bf_bits(float f) {
    union { float f; unsigned u; } x; x.f = f;
    unsigned r = x.u + 0x7FFFu + ((x.u >> 16) & 1u);   // round-to-nearest-even
    return (unsigned short)(r >> 16);
}
__device__ __forceinline__ float bf2f_bits(unsigned short h) {
    union { unsigned u; float f; } x; x.u = ((unsigned)h) << 16; return x.f;
}

// Load a 16-element bf16 fragment as two 16-byte chunks (ds/global *_b128 x2).
__device__ __forceinline__ bf16x16 ld_frag(const unsigned short* p0,
                                           const unsigned short* p1) {
    union { bf16x16 v; uint4 q[2]; } u;
    u.q[0] = *(const uint4*)p0;
    u.q[1] = *(const uint4*)p1;
    return u.v;
}

// CDNA5 async global->LDS copy (16B per lane), tracked by ASYNCcnt.
// Generic pointers to LDS keep the LDS byte offset in their low 32 bits.
__device__ __forceinline__ unsigned lds_addr32(const void* p) {
    return (unsigned)(uintptr_t)p;
}
__device__ __forceinline__ void async_g2lds_b128(unsigned lds_off, const void* gptr) {
    asm volatile("global_load_async_to_lds_b128 %0, %1, off"
                 :: "v"(lds_off), "v"(gptr) : "memory");
}
__device__ __forceinline__ void wait_async0() {
    asm volatile("s_wait_asynccnt 0x0" ::: "memory");
}

// ---------------------------------------------------------------------------
// fp32 -> bf16 elementwise (4 elements / thread)
// ---------------------------------------------------------------------------
__global__ void cvt_f32_to_bf16(const float* __restrict__ in,
                                unsigned short* __restrict__ out, int n4) {
    int i = blockIdx.x * blockDim.x + threadIdx.x;
    if (i >= n4) return;
    const float4 f = ((const float4*)in)[i];
    union { uint2 u; unsigned short h[4]; } o;
    o.h[0] = f2bf_bits(f.x); o.h[1] = f2bf_bits(f.y);
    o.h[2] = f2bf_bits(f.z); o.h[3] = f2bf_bits(f.w);
    ((uint2*)out)[i] = o.u;
}

// ---------------------------------------------------------------------------
// Weight transpose + convert: W fp32 [K=3072][N=3072] -> Wt bf16 [N][K]
// (so the WMMA B-fragment reads K values of one output column as two contiguous 8-element runs)
// ---------------------------------------------------------------------------
__global__ __launch_bounds__(256)
void transp_cvt(const float* __restrict__ W, unsigned short* __restrict__ Wt) {
    __shared__ float tile[32][33];
    const int tx = threadIdx.x, ty = threadIdx.y;
    const int nb = blockIdx.x * 32, kb = blockIdx.y * 32;
#pragma unroll
    for (int i = 0; i < 4; i++)
        tile[ty + 8 * i][tx] = W[(size_t)(kb + ty + 8 * i) * D_N + nb + tx];
    __syncthreads();
#pragma unroll
    for (int i = 0; i < 4; i++)
        Wt[(size_t)(nb + ty + 8 * i) * D_N + kb + tx] = f2bf_bits(tile[tx][ty + 8 * i]);
}

// ---------------------------------------------------------------------------
// WMMA GEMM: C[M,3072] = A[M,3072](bf16) @ W (Bt = W^T bf16 [N][K]) + bias
// Block tile 128x128, 8 waves (2M x 4N), wave tile 64x32 (4x2 WMMA accums),
// K-step 32. Double-buffered LDS filled by async global->LDS copies so the
// next tile streams in while WMMAs run on the current one.
// ---------------------------------------------------------------------------
#define LSTR 40   // LDS row stride in bf16 elements (80B, 16B aligned, conflict-spread)

template <bool HAS_BIAS, bool OUT_BF16>
__global__ __launch_bounds__(256)
void gemm_wmma_bf16(const unsigned short* __restrict__ A,
                    const unsigned short* __restrict__ Bt,
                    const float* __restrict__ bias,
                    float* __restrict__ Cf,
                    unsigned short* __restrict__ Cb) {
    __shared__ __align__(16) unsigned short As[2][128 * LSTR];
    __shared__ __align__(16) unsigned short Bs[2][128 * LSTR];

    const int t    = threadIdx.x;
    const int lane = t & 31;
    const int wave = t >> 5;
    const int lm   = lane & 15;
    const int half = lane >> 4;
    const int mblk = blockIdx.x * 128, nblk = blockIdx.y * 128;
    const int mw   = (wave & 1) * 64,  nw   = (wave >> 1) * 32;

    f32x8 acc[4][2];
#pragma unroll
    for (int mi = 0; mi < 4; mi++)
#pragma unroll
        for (int ni = 0; ni < 2; ni++) {
            f32x8 z = {0.f, 0.f, 0.f, 0.f, 0.f, 0.f, 0.f, 0.f};
            acc[mi][ni] = z;
        }

    const int row = t >> 1, seg = t & 1;     // 2 threads/row, 32B each
    const unsigned short* agp = A  + (size_t)(mblk + row) * D_N + seg * 16;
    const unsigned short* bgp = Bt + (size_t)(nblk + row) * D_N + seg * 16;
    unsigned aL[2], bL[2];
#pragma unroll
    for (int b = 0; b < 2; b++) {
        aL[b] = lds_addr32(&As[b][row * LSTR + seg * 16]);
        bL[b] = lds_addr32(&Bs[b][row * LSTR + seg * 16]);
    }

    // prologue: start streaming tile 0 into buffer 0
    async_g2lds_b128(aL[0],      agp);
    async_g2lds_b128(aL[0] + 16, agp + 8);
    async_g2lds_b128(bL[0],      bgp);
    async_g2lds_b128(bL[0] + 16, bgp + 8);
    agp += 32; bgp += 32;

    for (int kb = 0; kb < D_N; kb += 32) {
        const int cur = (kb >> 5) & 1;
        wait_async0();          // this wave's fills of buffer `cur` are done
        __syncthreads();        // ... and so is everyone else's
        if (kb + 32 < D_N) {    // stream next tile into the other buffer
            const int nxt = cur ^ 1;
            async_g2lds_b128(aL[nxt],      agp);
            async_g2lds_b128(aL[nxt] + 16, agp + 8);
            async_g2lds_b128(bL[nxt],      bgp);
            async_g2lds_b128(bL[nxt] + 16, bgp + 8);
            agp += 32; bgp += 32;
        }

        bf16x16 af[4], bfr[2];
#pragma unroll
        for (int mi = 0; mi < 4; mi++) {
            const unsigned short* rp = &As[cur][(mw + 16 * mi + lm) * LSTR];
            af[mi] = ld_frag(rp + half * 8, rp + 16 + half * 8);
        }
#pragma unroll
        for (int ni = 0; ni < 2; ni++) {
            const unsigned short* rp = &Bs[cur][(nw + 16 * ni + lm) * LSTR];
            bfr[ni] = ld_frag(rp + half * 8, rp + 16 + half * 8);
        }
#pragma unroll
        for (int mi = 0; mi < 4; mi++)
#pragma unroll
            for (int ni = 0; ni < 2; ni++)
                acc[mi][ni] = __builtin_amdgcn_wmma_f32_16x16x32_bf16(
                    false, af[mi], false, bfr[ni], (short)0, acc[mi][ni], false, false);
    }

#pragma unroll
    for (int mi = 0; mi < 4; mi++)
#pragma unroll
        for (int ni = 0; ni < 2; ni++) {
            const int col = nblk + nw + 16 * ni + lm;
            float bv = 0.f;
            if (HAS_BIAS) bv = bias[col];
#pragma unroll
            for (int r = 0; r < 8; r++) {
                const int rw = mblk + mw + 16 * mi + r + 8 * half;
                const float v = acc[mi][ni][r] + bv;
                if (OUT_BF16) Cb[(size_t)rw * D_N + col] = f2bf_bits(v);
                else          Cf[(size_t)rw * D_N + col] = v;
            }
        }
}

// ---------------------------------------------------------------------------
// Fused per-head RMSNorm (over DH=128) + interleaved RoPE, in place on bf16.
// One wave per (s, h) row; rows < S_TXT use the text-stream norm weight.
// oscale folds the attention 1/sqrt(DH) into Q (1.0 for K).
// ---------------------------------------------------------------------------
__global__ __launch_bounds__(256)
void rmsnorm_rope(unsigned short* __restrict__ buf,
                  const float* __restrict__ w_txt,
                  const float* __restrict__ w_img,
                  const float* __restrict__ cosT,
                  const float* __restrict__ sinT,
                  float oscale) {
    const int warp = blockIdx.x * 8 + (threadIdx.x >> 5);
    const int lane = threadIdx.x & 31;
    const int s  = warp / H_N;
    const int hh = warp % H_N;
    unsigned short* row = buf + (size_t)s * D_N + hh * DH_N;
    const int dh = lane * 4;

    union { uint2 u; unsigned short h[4]; } in;
    in.u = *(const uint2*)(row + dh);
    const float x0 = bf2f_bits(in.h[0]), x1 = bf2f_bits(in.h[1]);
    const float x2 = bf2f_bits(in.h[2]), x3 = bf2f_bits(in.h[3]);

    float ss = x0 * x0 + x1 * x1 + x2 * x2 + x3 * x3;
#pragma unroll
    for (int m = 1; m < 32; m <<= 1) ss += __shfl_xor(ss, m, 32);
    const float rinv = rsqrtf(ss * (1.f / 128.f) + 1e-5f) * oscale;

    const float* w = (s < S_TXT_N) ? w_txt : w_img;
    const float y0 = x0 * rinv * w[dh + 0], y1 = x1 * rinv * w[dh + 1];
    const float y2 = x2 * rinv * w[dh + 2], y3 = x3 * rinv * w[dh + 3];

    const float4 c  = *(const float4*)(cosT + (size_t)s * DH_N + dh);
    const float4 sn = *(const float4*)(sinT + (size_t)s * DH_N + dh);
    const float z0 = y0 * c.x - y1 * sn.x;
    const float z1 = y1 * c.y + y0 * sn.y;
    const float z2 = y2 * c.z - y3 * sn.z;
    const float z3 = y3 * c.w + y2 * sn.w;

    union { uint2 u; unsigned short h[4]; } o;
    o.h[0] = f2bf_bits(z0); o.h[1] = f2bf_bits(z1);
    o.h[2] = f2bf_bits(z2); o.h[3] = f2bf_bits(z3);
    *(uint2*)(row + dh) = o.u;
}

// ---------------------------------------------------------------------------
// Flash attention, one (q-tile=128, head) per block, 8 waves x 16 q-rows.
// Key chunk = 64. Q fragments live in registers for the whole loop (scale
// pre-folded by rmsnorm_rope); K chunk staged via async global->LDS (B-frag
// contiguous in DH); V staged transposed via ds stores; P goes through
// wave-private LDS (same-wave DS order).
// ---------------------------------------------------------------------------
#define CK      64
#define KS_STR 136   // 128 + 8 pad, 272B rows (16B aligned)
#define VT_STR  72   // 64 + 8 pad, 144B rows
#define P_STR   72

__global__ __launch_bounds__(256)
void flash_attn(const unsigned short* __restrict__ Qb,
                const unsigned short* __restrict__ Kb,
                const unsigned short* __restrict__ Vb,
                unsigned short* __restrict__ Ob) {
    __shared__ __align__(16) unsigned short Ks[CK * KS_STR];
    __shared__ __align__(16) unsigned short Vt[DH_N * VT_STR];
    __shared__ __align__(16) unsigned short Ps[8 * 16 * P_STR];

    const int t    = threadIdx.x;
    const int lane = t & 31;
    const int wave = t >> 5;
    const int lm   = lane & 15;
    const int half = lane >> 4;
    const int h    = blockIdx.y;
    const int qbase = blockIdx.x * 128 + wave * 16;

    // Q fragments (A layout); 1/sqrt(DH) already folded in upstream.
    bf16x16 qf[4];
    {
        const unsigned short* qrow = Qb + (size_t)(qbase + lm) * D_N + h * DH_N;
#pragma unroll
        for (int ks = 0; ks < 4; ks++)
            qf[ks] = ld_frag(qrow + 32 * ks + half * 8,
                             qrow + 32 * ks + 16 + half * 8);
    }

    float m_i[8], l_i[8];
    f32x8 o[8];
#pragma unroll
    for (int r = 0; r < 8; r++) { m_i[r] = -1e30f; l_i[r] = 0.f; }
#pragma unroll
    for (int j = 0; j < 8; j++) { f32x8 z = {0,0,0,0,0,0,0,0}; o[j] = z; }

    unsigned short* Prow = Ps + wave * 16 * P_STR;

    const int key = t >> 2, dseg = (t & 3) * 32;
    const unsigned kls = lds_addr32(&Ks[key * KS_STR + dseg]);

    for (int kc = 0; kc < S_TOT_N; kc += CK) {
        __syncthreads();
        {   // stage K (async, row-major) and V (transposed) for this chunk
            const unsigned short* ksrc = Kb + (size_t)(kc + key) * D_N + h * DH_N + dseg;
            async_g2lds_b128(kls,      ksrc);
            async_g2lds_b128(kls + 16, ksrc + 8);
            async_g2lds_b128(kls + 32, ksrc + 16);
            async_g2lds_b128(kls + 48, ksrc + 24);
            union { uint4 q[4]; unsigned short hh[32]; } vv;
            const uint4* vsrc = (const uint4*)(Vb + (size_t)(kc + key) * D_N + h * DH_N + dseg);
            vv.q[0] = vsrc[0]; vv.q[1] = vsrc[1]; vv.q[2] = vsrc[2]; vv.q[3] = vsrc[3];
#pragma unroll
            for (int j = 0; j < 32; j++)
                Vt[(dseg + j) * VT_STR + key] = vv.hh[j];
        }
        wait_async0();
        __syncthreads();

        // S = Q @ K^T : 4 key tiles x 4 DH steps = 16 WMMAs
        f32x8 sc[4];
#pragma unroll
        for (int ni = 0; ni < 4; ni++) {
            f32x8 z = {0,0,0,0,0,0,0,0}; sc[ni] = z;
#pragma unroll
            for (int ks = 0; ks < 4; ks++) {
                const unsigned short* rp = Ks + (16 * ni + lm) * KS_STR + 32 * ks + half * 8;
                const bf16x16 bk = ld_frag(rp, rp + 16);
                sc[ni] = __builtin_amdgcn_wmma_f32_16x16x32_bf16(
                    false, qf[ks], false, bk, (short)0, sc[ni], false, false);
            }
        }

        // Online softmax; each lane owns rows r+8*half, replicated over 16 lanes.
#pragma unroll
        for (int r = 0; r < 8; r++) {
            float mx = fmaxf(fmaxf(sc[0][r], sc[1][r]), fmaxf(sc[2][r], sc[3][r]));
#pragma unroll
            for (int m = 1; m < 16; m <<= 1) mx = fmaxf(mx, __shfl_xor(mx, m, 32));
            const float mnew  = fmaxf(m_i[r], mx);
            const float alpha = __expf(m_i[r] - mnew);
            m_i[r] = mnew;
            l_i[r] *= alpha;
#pragma unroll
            for (int j = 0; j < 8; j++) o[j][r] *= alpha;
            float rs = 0.f;
#pragma unroll
            for (int ni = 0; ni < 4; ni++) {
                const float p = __expf(sc[ni][r] - mnew);
                sc[ni][r] = p;
                rs += p;
            }
#pragma unroll
            for (int m = 1; m < 16; m <<= 1) rs += __shfl_xor(rs, m, 32);
            l_i[r] += rs;
#pragma unroll
            for (int ni = 0; ni < 4; ni++)   // C layout -> wave-private LDS
                Prow[(r + 8 * half) * P_STR + 16 * ni + lm] = f2bf_bits(sc[ni][r]);
        }

        // O += P @ V : 2 key steps x 8 DH tiles = 16 WMMAs
#pragma unroll
        for (int ks2 = 0; ks2 < 2; ks2++) {
            const unsigned short* pr = Prow + lm * P_STR + 32 * ks2;
            const bf16x16 pa = ld_frag(pr + half * 8, pr + 16 + half * 8);
#pragma unroll
            for (int nj = 0; nj < 8; nj++) {
                const unsigned short* vr = Vt + (16 * nj + lm) * VT_STR + 32 * ks2 + half * 8;
                const bf16x16 vb = ld_frag(vr, vr + 16);
                o[nj] = __builtin_amdgcn_wmma_f32_16x16x32_bf16(
                    false, pa, false, vb, (short)0, o[nj], false, false);
            }
        }
    }

    // Normalize and store bf16 joint output [s][h*128+dh]
#pragma unroll
    for (int r = 0; r < 8; r++) {
        const float inv = 1.f / l_i[r];
        const size_t rowoff = (size_t)(qbase + r + 8 * half) * D_N + h * DH_N;
#pragma unroll
        for (int nj = 0; nj < 8; nj++)
            Ob[rowoff + 16 * nj + lm] = f2bf_bits(o[nj][r] * inv);
    }
}

// ---------------------------------------------------------------------------
// Host orchestration
// ---------------------------------------------------------------------------
extern "C" void kernel_launch(void* const* d_in, const int* in_sizes, int n_in,
                              void* d_out, int out_size, void* d_ws, size_t ws_size,
                              hipStream_t stream) {
    const float* hidden       = (const float*)d_in[0];
    const float* enc          = (const float*)d_in[1];
    const float* rcos         = (const float*)d_in[2];
    const float* rsin         = (const float*)d_in[3];
    const float* wq           = (const float*)d_in[4];
    const float* wk           = (const float*)d_in[5];
    const float* wv           = (const float*)d_in[6];
    const float* norm_q_w     = (const float*)d_in[7];
    const float* norm_k_w     = (const float*)d_in[8];
    const float* add_wq       = (const float*)d_in[9];
    const float* add_bq       = (const float*)d_in[10];
    const float* add_wk       = (const float*)d_in[11];
    const float* add_bk       = (const float*)d_in[12];
    const float* add_wv       = (const float*)d_in[13];
    const float* add_bv       = (const float*)d_in[14];
    const float* norm_aq_w    = (const float*)d_in[15];
    const float* norm_ak_w    = (const float*)d_in[16];
    const float* to_out_w     = (const float*)d_in[17];
    const float* to_out_b     = (const float*)d_in[18];
    const float* to_add_out_w = (const float*)d_in[19];
    const float* to_add_out_b = (const float*)d_in[20];
    (void)in_sizes; (void)n_in; (void)out_size; (void)ws_size;

    char* ws = (char*)d_ws;
    size_t off = 0;
    auto alloc = [&](size_t bytes) -> unsigned short* {
        unsigned short* p = (unsigned short*)(ws + off);
        off += (bytes + 255) & ~(size_t)255;
        return p;
    };
    unsigned short* Xi = alloc((size_t)S_IMG_N * D_N * 2);
    unsigned short* Xt = alloc((size_t)S_TXT_N * D_N * 2);
    unsigned short* Wt[8];
    const float* Wsrc[8] = {wq, wk, wv, add_wq, add_wk, add_wv, to_out_w, to_add_out_w};
    for (int i = 0; i < 8; i++) Wt[i] = alloc((size_t)D_N * D_N * 2);
    unsigned short* Qb = alloc((size_t)S_TOT_N * D_N * 2);
    unsigned short* Kb = alloc((size_t)S_TOT_N * D_N * 2);
    unsigned short* Vb = alloc((size_t)S_TOT_N * D_N * 2);
    unsigned short* Ob = alloc((size_t)S_TOT_N * D_N * 2);

    // 1) activation + weight conversion (weights transposed to [N][K] bf16)
    cvt_f32_to_bf16<<<(S_IMG_N * D_N / 4 + 255) / 256, 256, 0, stream>>>(hidden, Xi, S_IMG_N * D_N / 4);
    cvt_f32_to_bf16<<<(S_TXT_N * D_N / 4 + 255) / 256, 256, 0, stream>>>(enc, Xt, S_TXT_N * D_N / 4);
    dim3 tb(32, 8), tg(D_N / 32, D_N / 32);
    for (int i = 0; i < 8; i++) transp_cvt<<<tg, tb, 0, stream>>>(Wsrc[i], Wt[i]);

    // 2) QKV projections into joint [2560,3072] buffers (text rows 0..511 first)
    const size_t txtRows = (size_t)S_TXT_N * D_N;
    dim3 g_img(S_IMG_N / 128, D_N / 128), g_txt(S_TXT_N / 128, D_N / 128);
    gemm_wmma_bf16<false, true><<<g_img, 256, 0, stream>>>(Xi, Wt[0], nullptr, nullptr, Qb + txtRows);
    gemm_wmma_bf16<false, true><<<g_img, 256, 0, stream>>>(Xi, Wt[1], nullptr, nullptr, Kb + txtRows);
    gemm_wmma_bf16<false, true><<<g_img, 256, 0, stream>>>(Xi, Wt[2], nullptr, nullptr, Vb + txtRows);
    gemm_wmma_bf16<true,  true><<<g_txt, 256, 0, stream>>>(Xt, Wt[3], add_bq, nullptr, Qb);
    gemm_wmma_bf16<true,  true><<<g_txt, 256, 0, stream>>>(Xt, Wt[4], add_bk, nullptr, Kb);
    gemm_wmma_bf16<true,  true><<<g_txt, 256, 0, stream>>>(Xt, Wt[5], add_bv, nullptr, Vb);

    // 3) per-head RMSNorm + RoPE on Q and K (joint positions); Q also gets
    //    the attention 1/sqrt(DH) folded in.
    const int nwarp = S_TOT_N * H_N;
    rmsnorm_rope<<<nwarp / 8, 256, 0, stream>>>(Qb, norm_aq_w, norm_q_w, rcos, rsin,
                                                0.08838834764831845f);
    rmsnorm_rope<<<nwarp / 8, 256, 0, stream>>>(Kb, norm_ak_w, norm_k_w, rcos, rsin, 1.0f);

    // 4) attention
    dim3 ga(S_TOT_N / 128, H_N);
    flash_attn<<<ga, 256, 0, stream>>>(Qb, Kb, Vb, Ob);

    // 5) output projections, fp32 straight into d_out (img_out then enc_out)
    float* outImg = (float*)d_out;
    float* outEnc = outImg + (size_t)S_IMG_N * D_N;
    gemm_wmma_bf16<true, false><<<g_img, 256, 0, stream>>>(Ob + txtRows, Wt[6], to_out_b, outImg, nullptr);
    gemm_wmma_bf16<true, false><<<g_txt, 256, 0, stream>>>(Ob, Wt[7], to_add_out_b, outEnc, nullptr);
}
